// SwinDepthController_79534204388072
// MI455X (gfx1250) — matrix operand order repaired and checked by reading, not compiled, not run on hardware
//
#include <hip/hip_runtime.h>

// ---------------------------------------------------------------------------
// Swin window attention for MI455X (gfx1250, wave32, WMMA bf16 path).
//   x[8192*49,128] -> qkv(bf16) -> per-(window,head) attention -> proj -> f32
// Matmuls use v_wmma_f32_16x16x32_bf16 with f32 accumulation; bf16->LDS
// staging uses GLOBAL_LOAD_ASYNC_TO_LDS (ASYNCcnt) when the toolchain has it.
// ---------------------------------------------------------------------------

typedef __bf16 bf16_t;
typedef __attribute__((ext_vector_type(16))) __bf16 v16bf;
typedef __attribute__((ext_vector_type(8)))  float  v8f;
typedef int v4i_vs __attribute__((vector_size(16)));   // 16B chunk for async copies

#define B_WIN   8192
#define NTOK    49
#define DIMC    128
#define HEADS   4
#define HD      32
#define NW      1024
#define MROWS   (B_WIN * NTOK)          /* 401408 = 3136*128 */
#define QSCALE  0.17677669529663687f    /* 1/sqrt(32) */

// --- async global->LDS path (gfx1250), guarded so we always compile --------
#define USE_ASYNC_LDS 0
#if defined(__has_builtin)
#if __has_builtin(__builtin_amdgcn_global_load_async_to_lds_b128) && \
    __has_builtin(__builtin_amdgcn_s_wait_asynccnt)
#undef USE_ASYNC_LDS
#define USE_ASYNC_LDS 1
#endif
#endif

#if USE_ASYNC_LDS
__device__ inline void async_copy16(const void* gsrc, void* ldst) {
  __builtin_amdgcn_global_load_async_to_lds_b128(
      (__attribute__((address_space(1))) v4i_vs*)gsrc,
      (__attribute__((address_space(3))) v4i_vs*)ldst, 0, 0);
}
#endif

union BF16x16 { v16bf v; bf16_t e[16]; uint4 q[2]; };

// ---- WMMA fragment loaders (contiguous k = 16*half + s in A and B, not the ISA table order, wave32) -------------
// A (16x32 bf16, MxK): lane m=lane&15 holds K = kb..kb+15, kb = 16*half
// (lane half). Two b128 LDS loads.
__device__ inline v16bf load_frag_a(const bf16_t* base, int ld, int row0,
                                    int k0, int lane) {
  const int m    = lane & 15;
  const int half = lane >> 4;
  const bf16_t* p = base + (row0 + m) * ld + k0;
  BF16x16 r;
  r.q[0] = *(const uint4*)(p + half * 16);
  r.q[1] = *(const uint4*)(p + half * 16 + 8);
  return r.v;
}

// B (32x16 bf16, KxN) with source stored "N-major, K contiguous" (i.e. rows
// of W / rows of K-matrix): lane n=lane&15 holds K = kb..kb+15, kb = 16*half.
__device__ inline v16bf load_frag_b(const bf16_t* base, int ld, int n0,
                                    int k0, int lane) {
  const int n  = lane & 15;
  const int kb = (lane >> 4) << 4;     // 0 or 16
  const bf16_t* p = base + (n0 + n) * ld + k0 + kb;
  BF16x16 r;
  r.q[0] = *(const uint4*)(p);
  r.q[1] = *(const uint4*)(p + 8);
  return r.v;
}

// B (32x16) with source stored K-major (V matrix [K rows][N cols]): strided.
__device__ inline v16bf load_frag_b_colmajor(const bf16_t* base, int ld,
                                             int k0, int n0, int lane) {
  const int n  = lane & 15;
  const int kb = (lane >> 4) << 4;
  const bf16_t* p = base + (size_t)(k0 + kb) * ld + n0 + n;
  BF16x16 r;
#pragma unroll
  for (int j = 0; j < 16; ++j) r.e[j] = p[j * ld];
  return r.v;
}

__device__ inline v8f wmma_bf16(v16bf a, v16bf b, v8f c) {
  return __builtin_amdgcn_wmma_f32_16x16x32_bf16(
      /*neg_a=*/false, a, /*neg_b=*/false, b,
      /*c_mod=*/(short)0, c, /*reuse_a=*/false, /*reuse_b=*/false);
}

// ---------------------------------------------------------------------------
// Kernel 1: qkv = x @ qkv_w^T + qkv_b  (q columns pre-scaled by 1/sqrt(hd))
// Block tile 128(M) x 64(N), K=128 fully staged in LDS as bf16. 256 thr/8 wv.
// (fp32 -> bf16 conversion forces the VGPR staging path here.)
// ---------------------------------------------------------------------------
__global__ __launch_bounds__(256) void qkv_gemm_kernel(
    const float* __restrict__ x, const float* __restrict__ w,
    const float* __restrict__ bvec, bf16_t* __restrict__ qkv_out) {
  constexpr int LDA = 136;  // 128 + 8 pad (keeps 16B alignment, avoids conflicts)
  __shared__ __align__(16) bf16_t Xs[128 * LDA];
  __shared__ __align__(16) bf16_t Ws[64 * LDA];

  const int tid  = threadIdx.x;
  const int row0 = blockIdx.x * 128;
  const int col0 = blockIdx.y * 64;

  // Stage X tile: 128x128 f32 -> bf16 (float4 loads).
  for (int i = tid; i < 128 * 32; i += 256) {
    int r = i >> 5, c4 = (i & 31) << 2;
    float4 f = *(const float4*)(x + (size_t)(row0 + r) * DIMC + c4);
    bf16_t* d = &Xs[r * LDA + c4];
    d[0] = (bf16_t)f.x; d[1] = (bf16_t)f.y; d[2] = (bf16_t)f.z; d[3] = (bf16_t)f.w;
  }
  // Stage W tile: rows col0..col0+63 of qkv_w[384][128].
  for (int i = tid; i < 64 * 32; i += 256) {
    int r = i >> 5, c4 = (i & 31) << 2;
    float4 f = *(const float4*)(w + (size_t)(col0 + r) * DIMC + c4);
    bf16_t* d = &Ws[r * LDA + c4];
    d[0] = (bf16_t)f.x; d[1] = (bf16_t)f.y; d[2] = (bf16_t)f.z; d[3] = (bf16_t)f.w;
  }
  __syncthreads();

  const int wv = tid >> 5, lane = tid & 31;
  v16bf a[4];
#pragma unroll
  for (int kt = 0; kt < 4; ++kt)
    a[kt] = load_frag_a(Xs, LDA, wv * 16, kt * 32, lane);

#pragma unroll
  for (int ct = 0; ct < 4; ++ct) {
    v8f acc = {};
#pragma unroll
    for (int kt = 0; kt < 4; ++kt) {
      v16bf bb = load_frag_b(Ws, LDA, ct * 16, kt * 32, lane);
      acc = wmma_bf16(a[kt], bb, acc);
    }
    const int cg   = col0 + ct * 16 + (lane & 15);
    const float bc = bvec[cg];
    const float sc = (cg < DIMC) ? QSCALE : 1.0f;   // scale q block
#pragma unroll
    for (int r = 0; r < 8; ++r) {
      int rowin = wv * 16 + r + ((lane >> 4) << 3);
      size_t m = (size_t)(row0 + rowin);
      qkv_out[m * 384 + cg] = (bf16_t)((acc[r] + bc) * sc);
    }
  }
}

// ---------------------------------------------------------------------------
// Kernel 2: per-(window, head) attention. grid = 8192*4 blocks, 128 thr/4 wv.
// S(64x64) = Q(64x32) K^T, +rel-pos-bias +shift-mask, softmax, O = P V.
// Q/K/V staging uses async global->LDS copies when available.
// ---------------------------------------------------------------------------
__global__ __launch_bounds__(128) void attn_kernel(
    const bf16_t* __restrict__ qkv, const float* __restrict__ rpb,
    const int* __restrict__ relidx, const float* __restrict__ mask,
    bf16_t* __restrict__ attn_out) {
  constexpr int LDS_S = 68;  // f32 S scratch leading dim
  constexpr int LDP   = 72;  // bf16 P leading dim (16B multiple)
  __shared__ __align__(16) bf16_t Qs[64 * 32];
  __shared__ __align__(16) bf16_t Ks[64 * 32];
  __shared__ __align__(16) bf16_t Vs[64 * 32];
  __shared__ __align__(16) float  Sf[64 * LDS_S];
  __shared__ __align__(16) bf16_t Ps[64 * LDP];

  const int tid = threadIdx.x;
  const int bid = blockIdx.x;
  const int win = bid >> 2;          // window index in [0, 8192)
  const int h   = bid & 3;           // head
  const int wi  = win & (NW - 1);    // window-in-image for mask

  const size_t base = (size_t)win * NTOK * 384 + (size_t)h * HD;

#if USE_ASYNC_LDS
  // Zero the pad rows (49..63) with regular ds stores.
  for (int i = tid; i < 15 * 32; i += 128) {
    int idx = NTOK * 32 + i;
    Qs[idx] = (bf16_t)0.0f; Ks[idx] = (bf16_t)0.0f; Vs[idx] = (bf16_t)0.0f;
  }
  // Async-copy the 49 valid rows: 4 x 16B chunks per row per matrix.
  for (int i = tid; i < NTOK * 4; i += 128) {
    int p = i >> 2, c8 = (i & 3) << 3;
    const bf16_t* s = qkv + base + (size_t)p * 384 + c8;
    async_copy16(s,       &Qs[p * 32 + c8]);
    async_copy16(s + 128, &Ks[p * 32 + c8]);
    async_copy16(s + 256, &Vs[p * 32 + c8]);
  }
  __builtin_amdgcn_s_wait_asynccnt(0);
  __syncthreads();
#else
  for (int i = tid; i < 64 * 32; i += 128) {
    int p = i >> 5, k = i & 31;
    bf16_t qv = (bf16_t)0.0f, kv = (bf16_t)0.0f, vv = (bf16_t)0.0f;
    if (p < NTOK) {
      const bf16_t* s = qkv + base + (size_t)p * 384 + k;
      qv = s[0]; kv = s[128]; vv = s[256];
    }
    Qs[i] = qv; Ks[i] = kv; Vs[i] = vv;
  }
  __syncthreads();
#endif

  const int wv   = tid >> 5, lane = tid & 31;
  const int ncol = lane & 15;
  const int hi8  = (lane >> 4) << 3;

  // S = Q K^T, one wmma (k=32) per 16x16 tile; wave wv owns row tile wv.
  {
    v16bf aq = load_frag_a(Qs, 32, wv * 16, 0, lane);
#pragma unroll
    for (int ct = 0; ct < 4; ++ct) {
      v16bf bk = load_frag_b(Ks, 32, ct * 16, 0, lane);
      v8f acc = {};
      acc = wmma_bf16(aq, bk, acc);
      const int col = ct * 16 + ncol;
#pragma unroll
      for (int r = 0; r < 8; ++r) {
        int row = wv * 16 + r + hi8;
        float v = acc[r];
        if (col < NTOK) {
          if (row < NTOK) {
            v += rpb[relidx[row * NTOK + col] * HEADS + h] +
                 mask[((size_t)wi * NTOK + row) * NTOK + col];
          }
        } else {
          v = -1e30f;   // mask padded key columns
        }
        Sf[row * LDS_S + col] = v;
      }
    }
  }
  __syncthreads();

  // Row softmax (rows 0..63; padded cols contribute exp(-1e30)≈0).
  if (tid < 64) {
    float* s = &Sf[tid * LDS_S];
    float mx = -1e30f;
    for (int c = 0; c < 64; ++c) mx = fmaxf(mx, s[c]);
    float sum = 0.0f;
    for (int c = 0; c < 64; ++c) { float e = __expf(s[c] - mx); s[c] = e; sum += e; }
    float inv = 1.0f / sum;
    bf16_t* pr = &Ps[tid * LDP];
    for (int c = 0; c < 64; ++c) pr[c] = (bf16_t)(s[c] * inv);
  }
  __syncthreads();

  // O = P(64x64) @ V(64x32): 2 N tiles x 2 k-steps per wave.
#pragma unroll
  for (int nt = 0; nt < 2; ++nt) {
    v8f acc = {};
#pragma unroll
    for (int kt = 0; kt < 2; ++kt) {
      v16bf ap = load_frag_a(Ps, LDP, wv * 16, kt * 32, lane);
      v16bf bv = load_frag_b_colmajor(Vs, 32, kt * 32, nt * 16, lane);
      acc = wmma_bf16(ap, bv, acc);
    }
    const int col = nt * 16 + ncol;
#pragma unroll
    for (int r = 0; r < 8; ++r) {
      int row = wv * 16 + r + hi8;
      if (row < NTOK) {
        attn_out[((size_t)win * NTOK + row) * DIMC + h * HD + col] = (bf16_t)acc[r];
      }
    }
  }
}

// ---------------------------------------------------------------------------
// Kernel 3: out = attn(bf16) @ proj_w^T + proj_b  (f32 output)
// A-tile staging is a straight bf16 copy -> async global->LDS when available.
// ---------------------------------------------------------------------------
__global__ __launch_bounds__(256) void proj_gemm_kernel(
    const bf16_t* __restrict__ a_in, const float* __restrict__ w,
    const float* __restrict__ bvec, float* __restrict__ out) {
  constexpr int LDA = 136;
  __shared__ __align__(16) bf16_t Xs[128 * LDA];
  __shared__ __align__(16) bf16_t Ws[64 * LDA];

  const int tid  = threadIdx.x;
  const int row0 = blockIdx.x * 128;
  const int col0 = blockIdx.y * 64;

  // Stage A tile (already bf16): 16B-chunk copies, async when available.
  for (int i = tid; i < 128 * 16; i += 256) {
    int r = i >> 4, c8 = (i & 15) << 3;
#if USE_ASYNC_LDS
    async_copy16(a_in + (size_t)(row0 + r) * DIMC + c8, &Xs[r * LDA + c8]);
#else
    *(uint4*)&Xs[r * LDA + c8] =
        *(const uint4*)(a_in + (size_t)(row0 + r) * DIMC + c8);
#endif
  }
  // Stage W tile f32 -> bf16 (conversion: VGPR path).
  for (int i = tid; i < 64 * 32; i += 256) {
    int r = i >> 5, c4 = (i & 31) << 2;
    float4 f = *(const float4*)(w + (size_t)(col0 + r) * DIMC + c4);
    bf16_t* d = &Ws[r * LDA + c4];
    d[0] = (bf16_t)f.x; d[1] = (bf16_t)f.y; d[2] = (bf16_t)f.z; d[3] = (bf16_t)f.w;
  }
#if USE_ASYNC_LDS
  __builtin_amdgcn_s_wait_asynccnt(0);
#endif
  __syncthreads();

  const int wv = tid >> 5, lane = tid & 31;
  v16bf a[4];
#pragma unroll
  for (int kt = 0; kt < 4; ++kt)
    a[kt] = load_frag_a(Xs, LDA, wv * 16, kt * 32, lane);

#pragma unroll
  for (int ct = 0; ct < 4; ++ct) {
    v8f acc = {};
#pragma unroll
    for (int kt = 0; kt < 4; ++kt) {
      v16bf bb = load_frag_b(Ws, LDA, ct * 16, kt * 32, lane);
      acc = wmma_bf16(a[kt], bb, acc);
    }
    const int cg   = col0 + ct * 16 + (lane & 15);
    const float bc = bvec[cg];
#pragma unroll
    for (int r = 0; r < 8; ++r) {
      int rowin = wv * 16 + r + ((lane >> 4) << 3);
      size_t m = (size_t)(row0 + rowin);
      out[m * DIMC + cg] = acc[r] + bc;
    }
  }
}

// ---------------------------------------------------------------------------
extern "C" void kernel_launch(void* const* d_in, const int* in_sizes, int n_in,
                              void* d_out, int out_size, void* d_ws,
                              size_t ws_size, hipStream_t stream) {
  (void)in_sizes; (void)n_in; (void)out_size; (void)ws_size;
  const float* x      = (const float*)d_in[0];
  const float* qkv_w  = (const float*)d_in[1];
  const float* qkv_b  = (const float*)d_in[2];
  const float* proj_w = (const float*)d_in[3];
  const float* proj_b = (const float*)d_in[4];
  const float* rpb    = (const float*)d_in[5];
  const float* mask   = (const float*)d_in[6];
  const int*   relidx = (const int*)d_in[7];

  bf16_t* qkv_ws  = (bf16_t*)d_ws;                       // [401408][384] bf16
  bf16_t* attn_ws = qkv_ws + (size_t)MROWS * 384;        // [401408][128] bf16
  float*  out     = (float*)d_out;

  dim3 g1(MROWS / 128, 384 / 64);
  qkv_gemm_kernel<<<g1, 256, 0, stream>>>(x, qkv_w, qkv_b, qkv_ws);

  attn_kernel<<<dim3(B_WIN * HEADS), 128, 0, stream>>>(qkv_ws, rpb, relidx,
                                                       mask, attn_ws);

  dim3 g3(MROWS / 128, 128 / 64);
  proj_gemm_kernel<<<g3, 256, 0, stream>>>(attn_ws, proj_w, proj_b, out);
}
